// Decoder_28509992911315
// MI455X (gfx1250) — matrix operand order repaired and checked by reading, not compiled, not run on hardware
//
#include <hip/hip_runtime.h>
#include <math.h>

// ---------------- problem constants (match reference) ----------------
#define N_LAYER 4
#define DIM     1024
#define NHEAD   16
#define DHEAD   64
#define VOCAB   32000
#define BATCH   2
#define SEQ     2048
#define MROWS   (BATCH * SEQ)   // 4096 token rows

// ---------------- CDNA5 WMMA types ----------------
typedef __attribute__((ext_vector_type(16))) __bf16 v16bf;
typedef __attribute__((ext_vector_type(8)))  float  v8f;

union FragU { unsigned int u[8]; v16bf v; };

__device__ __forceinline__ unsigned short f2bf(float f) {
    unsigned int u = __float_as_uint(f);
    unsigned int r = (u + 0x7FFFu + ((u >> 16) & 1u)) >> 16;   // RNE
    return (unsigned short)r;
}
__device__ __forceinline__ unsigned int pack2bf(float lo, float hi) {
    return (unsigned int)f2bf(lo) | ((unsigned int)f2bf(hi) << 16);
}
__device__ __forceinline__ v8f v8f_zero() {
    v8f z;
#pragma unroll
    for (int e = 0; e < 8; ++e) z[e] = 0.0f;
    return z;
}

// ---------------- 1) embedding + sinusoidal positional encoding ----------------
__global__ __launch_bounds__(256) void embed_pe_kernel(
    const int* __restrict__ ids, const float* __restrict__ embed, float* __restrict__ x) {
    int bl  = blockIdx.x;            // 0 .. M-1
    int l   = bl % SEQ;
    int tok = ids[bl];
    const float* e  = embed + (size_t)tok * DIM;
    float*       xo = x + (size_t)bl * DIM;
    const float c = -9.210340371976184f / (float)DIM;   // -ln(10000)/D
    for (int d = threadIdx.x; d < DIM; d += 256) {
        float freq = __expf((float)(d & ~1) * c);
        float ang  = (float)l * freq;
        float pe   = (d & 1) ? __cosf(ang) : __sinf(ang);
        xo[d] = e[d] + pe;
    }
}

// ---------------- 2) RMSNorm (one block per row) ----------------
__global__ __launch_bounds__(256) void rmsnorm_kernel(
    const float* __restrict__ x, const float* __restrict__ gamma, float* __restrict__ y) {
    __shared__ float red[256];
    int row = blockIdx.x;
    const float* xr = x + (size_t)row * DIM;
    float*       yr = y + (size_t)row * DIM;
    float vals[4];
    float s = 0.0f;
#pragma unroll
    for (int i = 0; i < 4; ++i) {
        float f = xr[threadIdx.x + i * 256];
        vals[i] = f; s += f * f;
    }
    red[threadIdx.x] = s;
    __syncthreads();
    for (int off = 128; off > 0; off >>= 1) {
        if (threadIdx.x < off) red[threadIdx.x] += red[threadIdx.x + off];
        __syncthreads();
    }
    float r = rsqrtf(red[0] / (float)DIM + 1e-6f);
#pragma unroll
    for (int i = 0; i < 4; ++i) {
        int d = threadIdx.x + i * 256;
        yr[d] = vals[i] * r * gamma[d];
    }
}

// ---------------- 3) bf16 WMMA GEMM: C = A[MxK] * W[KxN] (+bias)(+resid) ----------------
#define BM 128
#define BN 128
#define BK 32
#define GEMM_BIAS  1
#define GEMM_RESID 2

__global__ __launch_bounds__(256) void gemm_bf16_wmma(
    const float* __restrict__ A, const float* __restrict__ W,
    const float* __restrict__ bias, const float* __restrict__ resid,
    float* __restrict__ C, int M, int N, int K, int flags) {
    __shared__ __align__(16) unsigned short As[BM][BK + 2];   // [m][k], pad -> stride 17 dwords
    __shared__ __align__(16) unsigned short Bs[BN][BK + 2];   // [n][k] (W transposed), K-contiguous

    int tid  = threadIdx.x;
    int wid  = tid >> 5, lane = tid & 31, l16 = lane & 15, half = lane >> 4;
    int m0   = blockIdx.y * BM, n0 = blockIdx.x * BN;
    int wm   = wid & 3,  wn = wid >> 2;          // 4x2 wave grid
    int mBase = wm * 32, nBase = wn * 64;        // each wave: 32x64 output

    v8f acc[2][4];
#pragma unroll
    for (int i = 0; i < 2; ++i)
#pragma unroll
        for (int t = 0; t < 4; ++t) acc[i][t] = v8f_zero();

    int arow = tid >> 3;             // 0..31 (A tile row group)
    int ak   = (tid & 7) * 4;        // 0..28 (A tile k offset)
    int bkr  = tid & 31;             // 0..31 (W tile k row)
    int bng  = (tid >> 5) * 16;      // 0..112 (W tile n group)

    for (int k0 = 0; k0 < K; k0 += BK) {
        __syncthreads();
        // A tile -> LDS (f32 -> bf16), rows arow + {0,32,64,96}
#pragma unroll
        for (int rr = 0; rr < 4; ++rr) {
            int r = arow + rr * 32;
            const float4 f = *(const float4*)(A + (size_t)(m0 + r) * K + k0 + ak);
            As[r][ak + 0] = f2bf(f.x); As[r][ak + 1] = f2bf(f.y);
            As[r][ak + 2] = f2bf(f.z); As[r][ak + 3] = f2bf(f.w);
        }
        // W tile -> LDS transposed: Bs[n][k]
#pragma unroll
        for (int qq = 0; qq < 4; ++qq) {
            const float4 f = *(const float4*)(W + (size_t)(k0 + bkr) * N + n0 + bng + qq * 4);
            Bs[bng + qq * 4 + 0][bkr] = f2bf(f.x); Bs[bng + qq * 4 + 1][bkr] = f2bf(f.y);
            Bs[bng + qq * 4 + 2][bkr] = f2bf(f.z); Bs[bng + qq * 4 + 3][bkr] = f2bf(f.w);
        }
        __syncthreads();
        if (k0 + BK < K) {   // global_prefetch_b8 of next K tiles
            __builtin_prefetch(A + (size_t)(m0 + arow) * K + k0 + BK + ak, 0, 1);
            __builtin_prefetch(W + (size_t)(k0 + BK + bkr) * N + n0 + bng, 0, 1);
        }
        // A fragments (16-bit 16x32 ISA layout: lanes0-15 K{0..7,16..23}, lanes16-31 +8)
        v16bf afr[2];
#pragma unroll
        for (int i2 = 0; i2 < 2; ++i2) {
            FragU fu;
            int row = mBase + i2 * 16 + l16;
#pragma unroll
            for (int i = 0; i < 8; ++i) {
                int kk = ((i < 4) ? 2 * i : 16 + 2 * (i - 4)) + (half ? 8 : 0);
                fu.u[i] = *(const unsigned int*)&As[row][kk];
            }
            afr[i2] = fu.v;
        }
        // B fragments (32x16: lanes0-15 K{0..7,16..23}, lanes16-31 +8, pairs per VGPR)
        v16bf bfr[4];
#pragma unroll
        for (int t = 0; t < 4; ++t) {
            FragU fu;
            int col = nBase + t * 16 + l16;
#pragma unroll
            for (int i = 0; i < 8; ++i) {
                int kk = ((i < 4) ? 2 * i : 16 + 2 * (i - 4)) + (half ? 8 : 0);
                fu.u[i] = *(const unsigned int*)&Bs[col][kk];
            }
            bfr[t] = fu.v;
        }
#pragma unroll
        for (int i2 = 0; i2 < 2; ++i2)
#pragma unroll
            for (int t = 0; t < 4; ++t)
                acc[i2][t] = __builtin_amdgcn_wmma_f32_16x16x32_bf16(
                    false, afr[i2], false, bfr[t], (short)0, acc[i2][t], false, false);
    }
    // epilogue: C-layout (VGPR r -> row r / r+8, col = lane&15)
#pragma unroll
    for (int i2 = 0; i2 < 2; ++i2) {
        int rbase = m0 + mBase + i2 * 16 + (half ? 8 : 0);
#pragma unroll
        for (int t = 0; t < 4; ++t) {
            int col = n0 + nBase + t * 16 + l16;
            float bb = (flags & GEMM_BIAS) ? bias[col] : 0.0f;
#pragma unroll
            for (int r = 0; r < 8; ++r) {
                size_t idx = (size_t)(rbase + r) * N + col;
                float val = acc[i2][t][r] + bb;
                if (flags & GEMM_RESID) val += resid[idx];
                C[idx] = val;
            }
        }
    }
}

// ---------------- 4) causal flash attention (bf16 WMMA QK^T and PV) ----------------
__global__ __launch_bounds__(256) void attn_kernel(
    const float* __restrict__ q, const float* __restrict__ k,
    const float* __restrict__ v, float* __restrict__ o) {
    __shared__ __align__(16) unsigned short Ks[32][DHEAD + 2];     // [kv][dh]
    __shared__ __align__(16) unsigned short Vt[DHEAD][32 + 2];     // [dh][kv] (V^T)
    __shared__ __align__(16) unsigned short Ps[8][16][32 + 2];     // per-wave P scratch

    int tid = threadIdx.x, wid = tid >> 5, lane = tid & 31, l16 = lane & 15, half = lane >> 4;
    const int nqb = SEQ / 128;
    int bid = blockIdx.x;
    int qb = bid % nqb, h = (bid / nqb) % NHEAD, b = bid / (nqb * NHEAD);
    int qrow0 = qb * 128;
    int myq = qrow0 + wid * 16;          // this wave's 16 q rows

    // Q A-fragments, scaled by 1/sqrt(dh)=0.125 (dh contiguous in global -> pair loads)
    v16bf qa[2];
    {
        int qm = myq + l16;
        const float* qp = q + ((size_t)(b * SEQ + qm) * NHEAD + h) * DHEAD;
#pragma unroll
        for (int c = 0; c < 2; ++c) {
            FragU fu;
#pragma unroll
            for (int i = 0; i < 8; ++i) {
                int kk = c * 32 + ((i < 4) ? 2 * i : 16 + 2 * (i - 4)) + (half ? 8 : 0);
                float2 f = *(const float2*)(qp + kk);
                fu.u[i] = pack2bf(f.x * 0.125f, f.y * 0.125f);
            }
            qa[c] = fu.v;
        }
    }

    float mrow[8], lrow[8];
#pragma unroll
    for (int r = 0; r < 8; ++r) { mrow[r] = -1e30f; lrow[r] = 0.0f; }
    v8f ot[4];
#pragma unroll
    for (int t = 0; t < 4; ++t) ot[t] = v8f_zero();

    int kvr = tid >> 3;              // 0..31  kv row loaded by this thread
    int dc  = (tid & 7) * 8;         // 0..56  dh chunk
    int jend = (qrow0 + 128) >> 5;   // causal upper bound for this q block

    for (int j = 0; j < jend; ++j) {
        __syncthreads();
        {   // stage K and V^T (f32 -> bf16)
            size_t base = ((size_t)(b * SEQ + j * 32 + kvr) * NHEAD + h) * DHEAD + dc;
            const float4 kf0 = *(const float4*)(k + base);
            const float4 kf1 = *(const float4*)(k + base + 4);
            const float4 vf0 = *(const float4*)(v + base);
            const float4 vf1 = *(const float4*)(v + base + 4);
            Ks[kvr][dc + 0] = f2bf(kf0.x); Ks[kvr][dc + 1] = f2bf(kf0.y);
            Ks[kvr][dc + 2] = f2bf(kf0.z); Ks[kvr][dc + 3] = f2bf(kf0.w);
            Ks[kvr][dc + 4] = f2bf(kf1.x); Ks[kvr][dc + 5] = f2bf(kf1.y);
            Ks[kvr][dc + 6] = f2bf(kf1.z); Ks[kvr][dc + 7] = f2bf(kf1.w);
            Vt[dc + 0][kvr] = f2bf(vf0.x); Vt[dc + 1][kvr] = f2bf(vf0.y);
            Vt[dc + 2][kvr] = f2bf(vf0.z); Vt[dc + 3][kvr] = f2bf(vf0.w);
            Vt[dc + 4][kvr] = f2bf(vf1.x); Vt[dc + 5][kvr] = f2bf(vf1.y);
            Vt[dc + 6][kvr] = f2bf(vf1.z); Vt[dc + 7][kvr] = f2bf(vf1.w);
        }
        __syncthreads();

        // skip fully-masked kv blocks (wave-uniform guard; syncthreads stay uniform)
        if (j * 32 <= myq) {
            // S = Q K^T (two k=32 chunks over dh)
            v8f s0 = v8f_zero(), s1 = v8f_zero();
#pragma unroll
            for (int c = 0; c < 2; ++c) {
                FragU fb0, fb1;
#pragma unroll
                for (int i = 0; i < 8; ++i) {
                    int kk = c * 32 + ((i < 4) ? 2 * i : 16 + 2 * (i - 4)) + (half ? 8 : 0);
                    fb0.u[i] = *(const unsigned int*)&Ks[l16][kk];
                    fb1.u[i] = *(const unsigned int*)&Ks[16 + l16][kk];
                }
                s0 = __builtin_amdgcn_wmma_f32_16x16x32_bf16(false, qa[c], false, fb0.v, (short)0, s0, false, false);
                s1 = __builtin_amdgcn_wmma_f32_16x16x32_bf16(false, qa[c], false, fb1.v, (short)0, s1, false, false);
            }
            // causal mask (select, no divergence)
            if (j * 32 + 31 > myq + 15) {
#pragma unroll
                for (int r = 0; r < 8; ++r) {
                    int row = myq + (half ? 8 : 0) + r;
                    int c0 = j * 32 + l16;
                    s0[r] = (c0 > row)      ? -1e30f : s0[r];
                    s1[r] = (c0 + 16 > row) ? -1e30f : s1[r];
                }
            }
            // online softmax (row spread over 16 lanes -> shfl_xor reductions, wave32)
#pragma unroll
            for (int r = 0; r < 8; ++r) {
                float sm = fmaxf(s0[r], s1[r]);
#pragma unroll
                for (int off = 1; off < 16; off <<= 1) sm = fmaxf(sm, __shfl_xor(sm, off, 32));
                float mnew  = fmaxf(mrow[r], sm);
                float alpha = __expf(mrow[r] - mnew);
                float p0 = __expf(s0[r] - mnew), p1 = __expf(s1[r] - mnew);
                float ps = p0 + p1;
#pragma unroll
                for (int off = 1; off < 16; off <<= 1) ps += __shfl_xor(ps, off, 32);
                lrow[r] = lrow[r] * alpha + ps;
                mrow[r] = mnew;
#pragma unroll
                for (int t = 0; t < 4; ++t) ot[t][r] *= alpha;
                int lr = (half ? 8 : 0) + r;                    // C-layout -> A-layout relay
                Ps[wid][lr][l16]      = f2bf(p0);
                Ps[wid][lr][16 + l16] = f2bf(p1);
            }
            // P as A-fragment (same wave; HW keeps DS in order, compiler waits dscnt)
            FragU fa;
#pragma unroll
            for (int i = 0; i < 8; ++i) {
                int kk = ((i < 4) ? 2 * i : 16 + 2 * (i - 4)) + (half ? 8 : 0);
                fa.u[i] = *(const unsigned int*)&Ps[wid][l16][kk];
            }
            // O += P V  (B-fragment from V^T, kv contiguous)
#pragma unroll
            for (int t = 0; t < 4; ++t) {
                FragU fb;
#pragma unroll
                for (int i = 0; i < 8; ++i) {
                    int kk = ((i < 4) ? 2 * i : 16 + 2 * (i - 4)) + (half ? 8 : 0);
                    fb.u[i] = *(const unsigned int*)&Vt[t * 16 + l16][kk];
                }
                ot[t] = __builtin_amdgcn_wmma_f32_16x16x32_bf16(false, fa.v, false, fb.v, (short)0, ot[t], false, false);
            }
        }
    }
    // normalize and store [b, l, h, dh] (== [b, l, d] after reshape)
#pragma unroll
    for (int r = 0; r < 8; ++r) {
        float inv = 1.0f / lrow[r];
        int row = myq + (half ? 8 : 0) + r;
        float* op = o + ((size_t)(b * SEQ + row) * NHEAD + h) * DHEAD;
#pragma unroll
        for (int t = 0; t < 4; ++t) op[t * 16 + l16] = ot[t][r] * inv;
    }
}

// ---------------- 5) silu(proj * up) ----------------
__global__ __launch_bounds__(256) void silu_mul_kernel(
    const float* __restrict__ p, const float* __restrict__ u, float* __restrict__ g, int n) {
    int i = blockIdx.x * 256 + threadIdx.x;
    if (i < n) {
        float z = p[i] * u[i];
        g[i] = z / (1.0f + __expf(-z));
    }
}

// ---------------- host orchestration ----------------
extern "C" void kernel_launch(void* const* d_in, const int* in_sizes, int n_in,
                              void* d_out, int out_size, void* d_ws, size_t ws_size,
                              hipStream_t stream) {
    (void)in_sizes; (void)n_in; (void)out_size; (void)ws_size;
    const int*   ids     = (const int*)  d_in[0];
    const float* embed   = (const float*)d_in[1];
    const float* Wq      = (const float*)d_in[2];
    const float* bq      = (const float*)d_in[3];
    const float* Wk      = (const float*)d_in[4];
    const float* bk      = (const float*)d_in[5];
    const float* Wv      = (const float*)d_in[6];
    const float* bv      = (const float*)d_in[7];
    const float* Wo      = (const float*)d_in[8];
    const float* bo      = (const float*)d_in[9];
    const float* Wproj   = (const float*)d_in[10];
    const float* bproj   = (const float*)d_in[11];
    const float* Wup     = (const float*)d_in[12];
    const float* bup     = (const float*)d_in[13];
    const float* Wdown   = (const float*)d_in[14];
    const float* bdown   = (const float*)d_in[15];
    const float* gammas  = (const float*)d_in[16];
    const float* Wlogits = (const float*)d_in[17];
    const float* blogits = (const float*)d_in[18];
    float* out = (float*)d_out;

    // workspace: 10*M*D floats (~168 MB)
    const size_t MD = (size_t)MROWS * DIM;
    float* x  = (float*)d_ws;
    float* xn = x  + MD;
    float* qb_ = xn + MD;
    float* kb_ = qb_ + MD;
    float* vb_ = kb_ + MD;
    float* ao  = vb_ + MD;
    float* t0  = ao  + MD;        // M x 2D
    float* t1  = t0  + 2 * MD;    // M x 2D

    dim3 blk(256);
    dim3 gD (DIM / BN,       MROWS / BM);   // N = 1024
    dim3 g2D(2 * DIM / BN,   MROWS / BM);   // N = 2048
    dim3 gV (VOCAB / BN,     MROWS / BM);   // N = 32000 (250 blocks)

    embed_pe_kernel<<<MROWS, blk, 0, stream>>>(ids, embed, x);

    for (int i = 0; i < N_LAYER; ++i) {
        const size_t oDD = (size_t)i * DIM * DIM;
        const size_t oD2 = (size_t)i * DIM * 2 * DIM;
        rmsnorm_kernel<<<MROWS, blk, 0, stream>>>(x, gammas + (size_t)(2 * i) * DIM, xn);
        gemm_bf16_wmma<<<gD, blk, 0, stream>>>(xn, Wq + oDD, bq + (size_t)i * DIM, nullptr, qb_, MROWS, DIM, DIM, GEMM_BIAS);
        gemm_bf16_wmma<<<gD, blk, 0, stream>>>(xn, Wk + oDD, bk + (size_t)i * DIM, nullptr, kb_, MROWS, DIM, DIM, GEMM_BIAS);
        gemm_bf16_wmma<<<gD, blk, 0, stream>>>(xn, Wv + oDD, bv + (size_t)i * DIM, nullptr, vb_, MROWS, DIM, DIM, GEMM_BIAS);
        attn_kernel<<<BATCH * NHEAD * (SEQ / 128), blk, 0, stream>>>(qb_, kb_, vb_, ao);
        gemm_bf16_wmma<<<gD, blk, 0, stream>>>(ao, Wo + oDD, bo + (size_t)i * DIM, x, x, MROWS, DIM, DIM, GEMM_BIAS | GEMM_RESID);

        rmsnorm_kernel<<<MROWS, blk, 0, stream>>>(x, gammas + (size_t)(2 * i + 1) * DIM, xn);
        gemm_bf16_wmma<<<g2D, blk, 0, stream>>>(xn, Wproj + oD2, bproj + (size_t)i * 2 * DIM, nullptr, t0, MROWS, 2 * DIM, DIM, GEMM_BIAS);
        gemm_bf16_wmma<<<g2D, blk, 0, stream>>>(xn, Wup   + oD2, bup   + (size_t)i * 2 * DIM, nullptr, t1, MROWS, 2 * DIM, DIM, GEMM_BIAS);
        int nff = (int)(MD * 2);
        silu_mul_kernel<<<(nff + 255) / 256, blk, 0, stream>>>(t0, t1, t0, nff);
        gemm_bf16_wmma<<<gD, blk, 0, stream>>>(t0, Wdown + (size_t)i * 2 * DIM * DIM, bdown + (size_t)i * DIM, x, x, MROWS, DIM, 2 * DIM, GEMM_BIAS | GEMM_RESID);
    }
    // logits: [4096 x 1024] x [1024 x 32000] + blogits -> d_out
    gemm_bf16_wmma<<<gV, blk, 0, stream>>>(x, Wlogits, blogits, nullptr, out, MROWS, VOCAB, DIM, GEMM_BIAS);
}
